// DeformConv2D_6433861009830
// MI455X (gfx1250) — matrix operand order repaired and checked by reading, not compiled, not run on hardware
//
#include <hip/hip_runtime.h>
#include <hip/hip_bf16.h>

#define KK    3
#define N_    8
#define CIN   64
#define COUT  64
#define H_    128
#define W_    128
#define HW    (H_*W_)
#define KTOT  (CIN*KK*KK)   // 576
#define NPIX  (N_*HW)       // 131072

typedef __attribute__((ext_vector_type(16))) __bf16 v16bf;
typedef __attribute__((ext_vector_type(8)))  float  v8f;

static __device__ __forceinline__ unsigned short f32_to_bf16_rne(float f) {
    union { float f; unsigned u; } v; v.f = f;
    unsigned u = v.u;
    u += 0x7FFFu + ((u >> 16) & 1u);   // round-to-nearest-even
    return (unsigned short)(u >> 16);
}
static __device__ __forceinline__ float blo(unsigned u) {
    return __uint_as_float(u << 16);
}
static __device__ __forceinline__ float bhi(unsigned u) {
    return __uint_as_float(u & 0xffff0000u);
}
// packed bf16x2 ops (VOP3P) via inline asm — no clang builtin for these
static __device__ __forceinline__ unsigned pk_mul_bf16(unsigned a, unsigned b) {
    unsigned d;
    asm("v_pk_mul_bf16 %0, %1, %2" : "=v"(d) : "v"(a), "v"(b));
    return d;
}
static __device__ __forceinline__ unsigned pk_fma_bf16(unsigned a, unsigned b, unsigned c) {
    unsigned d;
    asm("v_pk_fma_bf16 %0, %1, %2, %3" : "=v"(d) : "v"(a), "v"(b), "v"(c));
    return d;
}

// ---------------------------------------------------------------------------
// Kernel 0: NCHW f32 -> NHWC bf16 staging copy (channel-contiguous gathers)
// ---------------------------------------------------------------------------
__global__ void to_nhwc_bf16(const float* __restrict__ inp,
                             unsigned short* __restrict__ nhwc) {
    int tid = threadIdx.x;
    int cq  = tid & 3;          // channel quarter (16 ch)
    int p   = tid >> 2;         // pixel within 64-pixel tile
    int pix = blockIdx.x * 64 + p;
    int n   = pix / HW;
    int rem = pix - n * HW;
    const float* ip = inp + ((size_t)n * CIN + cq * 16) * HW + rem;

    unsigned pk[8];
#pragma unroll
    for (int j = 0; j < 8; ++j) {
        unsigned short a = f32_to_bf16_rne(ip[(size_t)(2 * j) * HW]);
        unsigned short b = f32_to_bf16_rne(ip[(size_t)(2 * j + 1) * HW]);
        pk[j] = (unsigned)a | ((unsigned)b << 16);
    }
    uint4* op = (uint4*)(nhwc + (size_t)pix * CIN + cq * 16);
    op[0] = make_uint4(pk[0], pk[1], pk[2], pk[3]);
    op[1] = make_uint4(pk[4], pk[5], pk[6], pk[7]);
}

// ---------------------------------------------------------------------------
// Kernel 1: offsets = einsum('nchw,oc->nohw') + b_grid, from NHWC bf16
// ---------------------------------------------------------------------------
__global__ void offset_kernel(const unsigned short* __restrict__ nhwc,
                              const float* __restrict__ w_grid,
                              const float* __restrict__ b_grid,
                              float* __restrict__ off_ws) {
    int pix = blockIdx.x * blockDim.x + threadIdx.x;
    if (pix >= NPIX) return;
    const uint4* ip = (const uint4*)(nhwc + (size_t)pix * CIN);

    float acc[18];
#pragma unroll
    for (int o = 0; o < 18; ++o) acc[o] = b_grid[o];

#pragma unroll
    for (int g = 0; g < 8; ++g) {           // 8 channels per uint4
        uint4 q = ip[g];
        unsigned uu[4] = {q.x, q.y, q.z, q.w};
#pragma unroll
        for (int h = 0; h < 4; ++h) {
            float v0 = blo(uu[h]);
            float v1 = bhi(uu[h]);
            int c = g * 8 + h * 2;
#pragma unroll
            for (int o = 0; o < 18; ++o)
                acc[o] = fmaf(v0, w_grid[o * CIN + c],
                          fmaf(v1, w_grid[o * CIN + c + 1], acc[o]));
        }
    }
    float* op = off_ws + (size_t)pix * 18;
#pragma unroll
    for (int o = 0; o < 18; ++o) op[o] = acc[o];
}

// ---------------------------------------------------------------------------
// Kernel 2: w_conv (co,ci,ky,kx) f32 -> Abf[co][tap*64+ci] bf16 (K-reordered)
// ---------------------------------------------------------------------------
__global__ void prep_weights(const float* __restrict__ w_conv,
                             unsigned short* __restrict__ Abf) {
    int id = blockIdx.x * blockDim.x + threadIdx.x;   // co*576 + tap*64 + ci
    if (id >= COUT * KTOT) return;
    int co = id / KTOT;
    int r  = id - co * KTOT;
    int tap = r >> 6;
    int ci  = r & 63;
    Abf[id] = f32_to_bf16_rne(w_conv[((size_t)(co * CIN + ci)) * 9 + tap]);
}

// ---------------------------------------------------------------------------
// Kernel 3: fused bilinear sampling + GEMM via v_wmma_f32_16x16x32_bf16
//   D(64 x 64pix) = A(64 x 576) * B(576 x 64pix), B sampled on the fly in LDS
//   sampling itself runs on v_pk_fma_bf16 (packed bf16 VOP3P)
// ---------------------------------------------------------------------------
__global__ __launch_bounds__(256, 2) void fused_deform_conv(
    const unsigned short* __restrict__ nhwc,
    const float* __restrict__ off_ws,
    const unsigned short* __restrict__ Abf,
    float* __restrict__ out)
{
    __shared__ int      offl[64 * 9 * 4];   // 4 clamped byte-offsets per (pixel,tap)
    __shared__ unsigned wpkl[64 * 9 * 4];   // 4 bilinear weights, bf16x2-broadcast
    __shared__ unsigned short Bl[2][64][40]; // [buf][pixel][k], row = 80B (padded)

    const int tid  = threadIdx.x;
    const int pix0 = blockIdx.x * 64;        // 64 consecutive pixels: same n, y
    const int n    = pix0 / HW;
    const int prem = pix0 - n * HW;
    const int y    = prem / W_;
    const int xb   = prem - y * W_;

    // ---- precompute bilinear taps: 64 pixels x 9 taps (once per WG) ----
    for (int idx = tid; idx < 64 * 9; idx += 256) {
        int p   = idx / 9;
        int tap = idx - p * 9;
        int ky = tap / 3, kx = tap - ky * 3;
        const float* op = off_ws + (size_t)(pix0 + p) * 18 + (ky * 6 + kx * 2);
        float offx = op[0], offy = op[1];
        // px = x + (kx-1) + offx*0.5*(W-1) ; py = y + (ky-1) + offy*0.5*(H-1)
        float px = (float)(xb + p + kx - 1) + offx * (0.5f * (W_ - 1));
        float py = (float)(y + ky - 1)      + offy * (0.5f * (H_ - 1));

        float x0f = floorf(px), y0f = floorf(py);
        float wx1 = px - x0f, wy1 = py - y0f;
        float wx0 = 1.f - wx1, wy0 = 1.f - wy1;
        int x0 = (int)x0f, y0 = (int)y0f;
        int x1 = x0 + 1,   y1 = y0 + 1;
        bool vx0 = (x0 >= 0) && (x0 < W_), vx1 = (x1 >= 0) && (x1 < W_);
        bool vy0 = (y0 >= 0) && (y0 < H_), vy1 = (y1 >= 0) && (y1 < H_);
        int xc0 = x0 < 0 ? 0 : (x0 > W_-1 ? W_-1 : x0);
        int xc1 = x1 < 0 ? 0 : (x1 > W_-1 ? W_-1 : x1);
        int yc0 = y0 < 0 ? 0 : (y0 > H_-1 ? H_-1 : y0);
        int yc1 = y1 < 0 ? 0 : (y1 > H_-1 ? H_-1 : y1);
        float w00 = (vx0 && vy0) ? wx0 * wy0 : 0.f;
        float w01 = (vx1 && vy0) ? wx1 * wy0 : 0.f;
        float w10 = (vx0 && vy1) ? wx0 * wy1 : 0.f;
        float w11 = (vx1 && vy1) ? wx1 * wy1 : 0.f;

        // byte offsets into NHWC bf16 image (channel 0)
        *(int4*)&offl[idx * 4] = make_int4(
            (yc0 * W_ + xc0) * (CIN * 2), (yc0 * W_ + xc1) * (CIN * 2),
            (yc1 * W_ + xc0) * (CIN * 2), (yc1 * W_ + xc1) * (CIN * 2));
        // weights broadcast into both bf16 halves
        *(uint4*)&wpkl[idx * 4] = make_uint4(
            0x10001u * f32_to_bf16_rne(w00), 0x10001u * f32_to_bf16_rne(w01),
            0x10001u * f32_to_bf16_rne(w10), 0x10001u * f32_to_bf16_rne(w11));
    }
    __syncthreads();

    // fill-stage assignment: 1 pixel, 8 contiguous channels per thread
    const int p_f = tid & 63;
    const int cg  = tid >> 6;          // 0..3
    // wave / fragment assignment
    const int wave  = tid >> 5;        // 0..7
    const int lane  = tid & 31;
    const int mtile = wave & 3;        // c_out tile (16 rows each)
    const int nsub  = wave >> 2;       // pixel half (32 pixels each)
    const int lm    = lane & 15;
    const int lh    = lane >> 4;

    v8f acc0 = {0.f, 0.f, 0.f, 0.f, 0.f, 0.f, 0.f, 0.f};
    v8f acc1 = {0.f, 0.f, 0.f, 0.f, 0.f, 0.f, 0.f, 0.f};

    const char* nb = (const char*)(nhwc + (size_t)n * HW * CIN);

    for (int kc = 0; kc < 18; ++kc) {              // K chunks of 32 (tap, c-half)
        const int buf   = kc & 1;
        const int tap   = kc >> 1;
        const int chalf = (kc & 1) * 32;

        // ---- A fragment (bf16 weights, L2-resident), 16-bit A 16x32, contiguous k order (not the ISA table order):
        // lanes 0-15: M=lane, K=0-15; lanes 16-31: M=lane-16, K=16-31
        union { uint4 u[2]; v16bf v; } afr;
        {
            int co = mtile * 16 + lm;
            int kb = lh * 16;
            const unsigned short* ap = Abf + (size_t)co * KTOT + kc * 32 + kb;
            afr.u[0] = *(const uint4*)(ap);
            afr.u[1] = *(const uint4*)(ap + 8);
        }

        // ---- produce B tile: packed-bf16 bilinear sampling -> LDS ----
        {
            const int idx = p_f * 9 + tap;
            int4  ofs = *(const int4*)&offl[idx * 4];
            uint4 wv  = *(const uint4*)&wpkl[idx * 4];

            const char* base = nb + (size_t)(chalf + cg * 8) * 2; // channel byte off
            uint4 q00 = *(const uint4*)(base + ofs.x);
            uint4 q01 = *(const uint4*)(base + ofs.y);
            uint4 q10 = *(const uint4*)(base + ofs.z);
            uint4 q11 = *(const uint4*)(base + ofs.w);
            unsigned u00[4] = {q00.x, q00.y, q00.z, q00.w};
            unsigned u01[4] = {q01.x, q01.y, q01.z, q01.w};
            unsigned u10[4] = {q10.x, q10.y, q10.z, q10.w};
            unsigned u11[4] = {q11.x, q11.y, q11.z, q11.w};

            unsigned rp[4];
#pragma unroll
            for (int h = 0; h < 4; ++h) {
                unsigned d = pk_mul_bf16(wv.x, u00[h]);
                d = pk_fma_bf16(wv.y, u01[h], d);
                d = pk_fma_bf16(wv.z, u10[h], d);
                d = pk_fma_bf16(wv.w, u11[h], d);
                rp[h] = d;                     // already a packed bf16 pair
            }
            *(uint4*)&Bl[buf][p_f][cg * 8] = make_uint4(rp[0], rp[1], rp[2], rp[3]);
        }
        __syncthreads();   // single barrier per chunk (ping-pong buffers)

        // ---- B fragments from LDS, 16-bit B 32x16 layout:
        // lanes 0-15: N=lane, K=0-15; lanes 16-31: N=lane-16, K=16-31
        union { uint4 u[2]; v16bf v; } bfr0, bfr1;
        {
            int kb2 = lh * 16;
            const unsigned short* bp0 = &Bl[buf][nsub * 32 + lm][kb2];
            bfr0.u[0] = *(const uint4*)bp0;
            bfr0.u[1] = *(const uint4*)(bp0 + 8);
            const unsigned short* bp1 = &Bl[buf][nsub * 32 + 16 + lm][kb2];
            bfr1.u[0] = *(const uint4*)bp1;
            bfr1.u[1] = *(const uint4*)(bp1 + 8);
        }

        acc0 = __builtin_amdgcn_wmma_f32_16x16x32_bf16(
            false, afr.v, false, bfr0.v, (short)0, acc0, false, false);
        acc1 = __builtin_amdgcn_wmma_f32_16x16x32_bf16(
            false, afr.v, false, bfr1.v, (short)0, acc1, false, false);
    }

    // ---- store D: 32-bit C/D 16x16 layout (VGPR r -> M=r / r+8 by lane half) ----
    {
        float* obase = out + (size_t)n * COUT * HW + (size_t)y * W_ + xb;
        int xl = nsub * 32 + lm;
#pragma unroll
        for (int r = 0; r < 8; ++r) {
            int co = mtile * 16 + r + lh * 8;
            obase[(size_t)co * HW + xl]      = acc0[r];
            obase[(size_t)co * HW + xl + 16] = acc1[r];
        }
    }
}

// ---------------------------------------------------------------------------
extern "C" void kernel_launch(void* const* d_in, const int* in_sizes, int n_in,
                              void* d_out, int out_size, void* d_ws, size_t ws_size,
                              hipStream_t stream) {
    (void)in_sizes; (void)n_in; (void)out_size; (void)ws_size;
    const float* inp    = (const float*)d_in[0];
    const float* w_grid = (const float*)d_in[1];
    const float* b_grid = (const float*)d_in[2];
    const float* w_conv = (const float*)d_in[3];

    // workspace layout
    char* ws = (char*)d_ws;
    float*          off_ws = (float*)ws;                              // 9.44 MB
    unsigned short* Abf    = (unsigned short*)(ws + (size_t)NPIX * 18 * sizeof(float));
    unsigned short* nhwc   = (unsigned short*)(ws + (size_t)NPIX * 18 * sizeof(float)
                                                  + (size_t)COUT * KTOT * sizeof(unsigned short));
                                                                      // +16.8 MB

    to_nhwc_bf16<<<dim3(NPIX / 64), dim3(256), 0, stream>>>(inp, nhwc);
    offset_kernel<<<dim3((NPIX + 255) / 256), dim3(256), 0, stream>>>(
        nhwc, w_grid, b_grid, off_ws);
    prep_weights<<<dim3((COUT * KTOT + 255) / 256), dim3(256), 0, stream>>>(
        w_conv, Abf);
    fused_deform_conv<<<dim3(NPIX / 64), dim3(256), 0, stream>>>(
        nhwc, off_ws, Abf, (float*)d_out);
}
